// Decoder_60241211294041
// MI455X (gfx1250) — matrix operand order repaired and checked by reading, not compiled, not run on hardware
//
#include <hip/hip_runtime.h>
#include <hip/hip_bf16.h>
#include <math.h>

typedef _Float16 half_t;
typedef __attribute__((ext_vector_type(16))) _Float16 v16h;
typedef __attribute__((ext_vector_type(8)))  _Float16 v8h;
typedef __attribute__((ext_vector_type(8)))  float    v8f;

#define B_    64
#define DCTX  684
#define DATT  512
#define HID   256
#define MEMB  256
#define HH_   16
#define WW_   48
#define P_    768     // H*W
#define KPAD  704     // 684 ctx + 9 patch + pad -> 22 chunks of 32
#define KCHUNKS 22

__device__ __forceinline__ float sigm_(float x){ return 1.0f/(1.0f+expf(-x)); }

// B16[a][k] fp16: k<684 -> Ua_w[a,k]; 684..692 -> Ufw[a,t]=sum_m Uf[a,m]*Qw[m,t]; else 0
__global__ __launch_bounds__(256) void k_prepB(
    const float* __restrict__ Ua_w, const float* __restrict__ Uf,
    const float* __restrict__ Qw, half_t* __restrict__ B16) {
  int idx = blockIdx.x*256 + threadIdx.x;
  if (idx >= DATT*KPAD) return;
  int a = idx / KPAD, k = idx % KPAD;
  float v = 0.0f;
  if (k < DCTX) {
    v = Ua_w[a*DCTX + k];
  } else if (k < DCTX + 9) {
    int t = k - DCTX;
    const float* uf = Uf + a*256;
    float s = 0.0f;
    for (int m = 0; m < 256; ++m) s += uf[m] * Qw[m*9 + t];
    v = s;
  }
  B16[idx] = (half_t)v;
}

__global__ __launch_bounds__(256) void k_embed(
    const int* __restrict__ y, const float* __restrict__ emb, float* __restrict__ embw) {
  int idx = blockIdx.x*256 + threadIdx.x;
  if (idx >= B_*MEMB) return;
  int b = idx / MEMB, m = idx % MEMB;
  embw[idx] = emb[(size_t)y[b]*MEMB + m];
}

__global__ __launch_bounds__(256) void k_gates(
    const float* __restrict__ x, int xdim,
    const float* __restrict__ h,
    const float* __restrict__ W_ih, const float* __restrict__ W_hh,
    const float* __restrict__ b_ih, const float* __restrict__ b_hh,
    float* __restrict__ gi, float* __restrict__ gh) {
  int idx = blockIdx.x*256 + threadIdx.x;
  if (idx >= B_*768) return;
  int b = idx / 768, j = idx % 768;
  const float* xb = x + (size_t)b*xdim;
  const float* hb = h + (size_t)b*HID;
  const float* wi = W_ih + (size_t)j*xdim;
  const float* wh = W_hh + (size_t)j*HID;
  float si = b_ih[j], sh = b_hh[j];
  for (int m = 0; m < xdim; ++m) si += xb[m]*wi[m];
  for (int m = 0; m < HID;  ++m) sh += hb[m]*wh[m];
  gi[idx] = si; gh[idx] = sh;
}

__global__ __launch_bounds__(256) void k_gru_combine(
    const float* __restrict__ gi, const float* __restrict__ gh,
    const float* __restrict__ hprev, const float* __restrict__ y_mask,
    float* __restrict__ snew) {
  int idx = blockIdx.x*256 + threadIdx.x;
  if (idx >= B_*HID) return;
  int b = idx / HID, hh = idx % HID;
  const float* gib = gi + (size_t)b*768;
  const float* ghb = gh + (size_t)b*768;
  float r = sigm_(gib[hh]       + ghb[hh]);
  float z = sigm_(gib[HID+hh]   + ghb[HID+hh]);
  float n = tanhf(gib[2*HID+hh] + r*ghb[2*HID+hh]);
  float hp = hprev[idx];
  float sv = (1.0f - z)*n + z*hp;
  float m = y_mask[b];
  snew[idx] = m*sv + (1.0f - m)*hp;
}

__global__ __launch_bounds__(256) void k_biasA(
    const float* __restrict__ sh, const float* __restrict__ Wa,
    const float* __restrict__ Ua_b, const float* __restrict__ Uf_b,
    float* __restrict__ biasA) {
  int idx = blockIdx.x*256 + threadIdx.x;
  if (idx >= B_*DATT) return;
  int b = idx / DATT, a = idx % DATT;
  const float* shb = sh + (size_t)b*HID;
  const float* wa  = Wa + (size_t)a*HID;
  float s = Ua_b[a] + Uf_b[a];
  for (int h = 0; h < HID; ++h) s += shb[h]*wa[h];
  biasA[idx] = s;
}

// fused attention-energy GEMM via WMMA f32_16x16x32_f16
__global__ __launch_bounds__(256) void k_attn(
    const float* __restrict__ ctx, const float* __restrict__ alpha_past,
    const half_t* __restrict__ B16, const float* __restrict__ biasA,
    const float* __restrict__ va, const float* __restrict__ va_b,
    float* __restrict__ alog) {
  __shared__ __align__(16) half_t Alds[32][32];   // [m][k]
  __shared__ float lred[32];

  const int b  = blockIdx.y;
  const int p0 = blockIdx.x * 32;
  const int t  = threadIdx.x;
  const int lane = t & 31;
  const int wv   = t >> 5;
  const int hi   = lane >> 4;
  const int ln16 = lane & 15;
  const int n_base = wv * 64;

  if (t < 32) lred[t] = 0.0f;

  v8f acc0[4] = {};
  v8f acc1[4] = {};

  const float* ctxb = ctx + (size_t)b*DCTX*P_ + p0;

  for (int kc = 0; kc < KCHUNKS; ++kc) {
    const int kg0 = kc * 32;
    {   // stage A tile (32 p x 32 k) as fp16 in LDS
      int kl = t >> 3;          // 0..31
      int pl = (t & 7) * 4;     // 0,4,...,28
      int kglob = kg0 + kl;
      float v0, v1, v2, v3;
      if (kglob < DCTX) {
        const float* src = ctxb + (size_t)kglob*P_ + pl;
        v0 = src[0]; v1 = src[1]; v2 = src[2]; v3 = src[3];
      } else if (kglob < DCTX + 9) {
        int tt = kglob - DCTX;
        int ky = tt/3 - 1, kx = tt%3 - 1;
        float vv[4];
        #pragma unroll
        for (int i = 0; i < 4; ++i) {
          int p  = p0 + pl + i;
          int hh = p / WW_ + ky;
          int ww = p % WW_ + kx;
          vv[i] = (hh >= 0 && hh < HH_ && ww >= 0 && ww < WW_)
                    ? alpha_past[(size_t)b*P_ + hh*WW_ + ww] : 0.0f;
        }
        v0 = vv[0]; v1 = vv[1]; v2 = vv[2]; v3 = vv[3];
      } else {
        v0 = v1 = v2 = v3 = 0.0f;
      }
      Alds[pl+0][kl] = (half_t)v0;
      Alds[pl+1][kl] = (half_t)v1;
      Alds[pl+2][kl] = (half_t)v2;
      Alds[pl+3][kl] = (half_t)v3;
    }
    __syncthreads();

    // A fragments per ISA 16-bit 16x32 layout
    v16h af0, af1;
    {
      const half_t* ar0 = &Alds[ln16][0];
      const half_t* ar1 = &Alds[16 + ln16][0];
      v8h a0lo = *(const v8h*)(ar0 + hi*8);
      v8h a0hi = *(const v8h*)(ar0 + 16 + hi*8);
      v8h a1lo = *(const v8h*)(ar1 + hi*8);
      v8h a1hi = *(const v8h*)(ar1 + 16 + hi*8);
      #pragma unroll
      for (int j = 0; j < 8; ++j) {
        af0[j] = a0lo[j]; af0[j+8] = a0hi[j];
        af1[j] = a1lo[j]; af1[j+8] = a1hi[j];
      }
    }

    #pragma unroll
    for (int nt = 0; nt < 4; ++nt) {
      int a = n_base + nt*16 + ln16;
      const half_t* brow = B16 + (size_t)a*KPAD + kg0 + hi*8;
      v8h blo = *(const v8h*)(brow);
      v8h bhi = *(const v8h*)(brow + 16);
      v16h bf;
      #pragma unroll
      for (int j = 0; j < 8; ++j) { bf[j] = blo[j]; bf[j+8] = bhi[j]; }
      acc0[nt] = __builtin_amdgcn_wmma_f32_16x16x32_f16(
                   false, af0, false, bf, (short)0, acc0[nt], false, false);
      acc1[nt] = __builtin_amdgcn_wmma_f32_16x16x32_f16(
                   false, af1, false, bf, (short)0, acc1[nt], false, false);
    }
    __syncthreads();
  }

  // epilogue: tanh + dot(va), reduce over a
  float part0[8], part1[8];
  #pragma unroll
  for (int r = 0; r < 8; ++r) { part0[r] = 0.0f; part1[r] = 0.0f; }

  #pragma unroll
  for (int nt = 0; nt < 4; ++nt) {
    int a = n_base + nt*16 + ln16;
    float bb  = biasA[(size_t)b*DATT + a];
    float vva = va[a];
    #pragma unroll
    for (int r = 0; r < 8; ++r) {
      part0[r] += tanhf(acc0[nt][r] + bb) * vva;
      part1[r] += tanhf(acc1[nt][r] + bb) * vva;
    }
  }
  #pragma unroll
  for (int off = 1; off < 16; off <<= 1) {
    #pragma unroll
    for (int r = 0; r < 8; ++r) {
      part0[r] += __shfl_xor(part0[r], off, 32);
      part1[r] += __shfl_xor(part1[r], off, 32);
    }
  }
  if (ln16 == 0) {
    int mb = 8*hi;
    #pragma unroll
    for (int r = 0; r < 8; ++r) {
      atomicAdd(&lred[mb + r],      part0[r]);
      atomicAdd(&lred[16 + mb + r], part1[r]);
    }
  }
  __syncthreads();
  if (t < 32) alog[(size_t)b*P_ + p0 + t] = lred[t] + va_b[0];
}

__global__ __launch_bounds__(256) void k_softmax_ct(
    const float* __restrict__ alog, const float* __restrict__ ctx,
    float* __restrict__ alpha_out, float* __restrict__ ct) {
  __shared__ float sa[P_];
  __shared__ float red[256];
  int b = blockIdx.x, t = threadIdx.x;
  float mx = -1e30f;
  for (int p = t; p < P_; p += 256) { float v = alog[(size_t)b*P_ + p]; sa[p] = v; mx = fmaxf(mx, v); }
  red[t] = mx; __syncthreads();
  for (int s = 128; s > 0; s >>= 1) { if (t < s) red[t] = fmaxf(red[t], red[t+s]); __syncthreads(); }
  mx = red[0]; __syncthreads();
  float sum = 0.0f;
  for (int p = t; p < P_; p += 256) { float e = expf(sa[p] - mx); sa[p] = e; sum += e; }
  red[t] = sum; __syncthreads();
  for (int s = 128; s > 0; s >>= 1) { if (t < s) red[t] += red[t+s]; __syncthreads(); }
  float inv = 1.0f / red[0];
  for (int p = t; p < P_; p += 256) { float av = sa[p]*inv; sa[p] = av; alpha_out[(size_t)b*P_ + p] = av; }
  __syncthreads();
  for (int d = t; d < DCTX; d += 256) {
    const float* row = ctx + (size_t)b*DCTX*P_ + (size_t)d*P_;
    float s = 0.0f;
    #pragma unroll 4
    for (int p = 0; p < P_; ++p) s += row[p]*sa[p];
    ct[(size_t)b*DCTX + d] = s;
  }
}

__global__ __launch_bounds__(256) void k_logit(
    const float* __restrict__ ct, const float* __restrict__ s,
    const float* __restrict__ embw,
    const float* __restrict__ Wct, const float* __restrict__ Wct_b,
    const float* __restrict__ Wht, const float* __restrict__ Wht_b,
    float* __restrict__ logit) {
  int idx = blockIdx.x*256 + threadIdx.x;
  if (idx >= B_*MEMB) return;
  int b = idx / MEMB, m = idx % MEMB;
  const float* ctb = ct + (size_t)b*DCTX;
  const float* sb  = s  + (size_t)b*HID;
  const float* wc  = Wct + (size_t)m*DCTX;
  const float* wh  = Wht + (size_t)m*HID;
  float acc = Wct_b[m] + Wht_b[m] + embw[idx];
  for (int d = 0; d < DCTX; ++d) acc += ctb[d]*wc[d];
  for (int h = 0; h < HID;  ++h) acc += sb[h]*wh[h];
  logit[idx] = acc;
}

__global__ __launch_bounds__(256) void k_out(
    const float* __restrict__ logit, const float* __restrict__ W0,
    const float* __restrict__ W0_b, float* __restrict__ out) {
  int idx = blockIdx.x*256 + threadIdx.x;
  if (idx >= B_*128) return;
  int b = idx / 128, k = idx % 128;
  const float* lb = logit + (size_t)b*MEMB;
  const float* w0 = W0 + (size_t)k*128;
  float s = W0_b[k];
  for (int i = 0; i < 128; ++i) s += fmaxf(lb[2*i], lb[2*i+1]) * w0[i];
  out[idx] = s;
}

extern "C" void kernel_launch(void* const* d_in, const int* in_sizes, int n_in,
                              void* d_out, int out_size, void* d_ws, size_t ws_size,
                              hipStream_t stream) {
  const int*   y        = (const int*)  d_in[0];
  const float* y_mask   = (const float*)d_in[1];
  const float* context  = (const float*)d_in[2];
  const float* pre_hid  = (const float*)d_in[3];
  const float* a_past   = (const float*)d_in[4];
  const float* emb      = (const float*)d_in[5];
  const float* W_ih1    = (const float*)d_in[6];
  const float* W_hh1    = (const float*)d_in[7];
  const float* b_ih1    = (const float*)d_in[8];
  const float* b_hh1    = (const float*)d_in[9];
  const float* Ua_w     = (const float*)d_in[10];
  const float* Ua_b     = (const float*)d_in[11];
  const float* Wa       = (const float*)d_in[12];
  const float* Qw       = (const float*)d_in[13];
  const float* Uf       = (const float*)d_in[14];
  const float* Uf_b     = (const float*)d_in[15];
  const float* va       = (const float*)d_in[16];
  const float* va_b     = (const float*)d_in[17];
  const float* W_ih2    = (const float*)d_in[18];
  const float* W_hh2    = (const float*)d_in[19];
  const float* b_ih2    = (const float*)d_in[20];
  const float* b_hh2    = (const float*)d_in[21];
  const float* Wct      = (const float*)d_in[22];
  const float* Wct_b    = (const float*)d_in[23];
  const float* Wht      = (const float*)d_in[24];
  const float* Wht_b    = (const float*)d_in[25];
  const float* W0       = (const float*)d_in[26];
  const float* W0_b     = (const float*)d_in[27];

  float* out       = (float*)d_out;           // [64,128]
  float* s_out     = out + B_*128;            // [64,256]
  float* alpha_out = s_out + B_*HID;          // [64,768]

  float* wsf = (float*)d_ws;
  size_t o = 0;
  float* embw  = wsf + o; o += B_*MEMB;
  float* sh    = wsf + o; o += B_*HID;
  float* gi    = wsf + o; o += B_*768;
  float* gh    = wsf + o; o += B_*768;
  float* biasA = wsf + o; o += B_*DATT;
  float* alog  = wsf + o; o += B_*P_;
  float* ct    = wsf + o; o += B_*DCTX;
  float* logit = wsf + o; o += B_*MEMB;
  o = (o + 15) & ~(size_t)15;
  half_t* B16  = (half_t*)(wsf + o);          // [512][704] fp16

  k_prepB<<<(DATT*KPAD + 255)/256, 256, 0, stream>>>(Ua_w, Uf, Qw, B16);
  k_embed<<<(B_*MEMB + 255)/256, 256, 0, stream>>>(y, emb, embw);
  k_gates<<<(B_*768 + 255)/256, 256, 0, stream>>>(embw, MEMB, pre_hid, W_ih1, W_hh1, b_ih1, b_hh1, gi, gh);
  k_gru_combine<<<(B_*HID + 255)/256, 256, 0, stream>>>(gi, gh, pre_hid, y_mask, sh);
  k_biasA<<<(B_*DATT + 255)/256, 256, 0, stream>>>(sh, Wa, Ua_b, Uf_b, biasA);
  {
    dim3 grid(P_/32, B_);
    k_attn<<<grid, 256, 0, stream>>>(context, a_past, B16, biasA, va, va_b, alog);
  }
  k_softmax_ct<<<B_, 256, 0, stream>>>(alog, context, alpha_out, ct);
  k_gates<<<(B_*768 + 255)/256, 256, 0, stream>>>(ct, DCTX, sh, W_ih2, W_hh2, b_ih2, b_hh2, gi, gh);
  k_gru_combine<<<(B_*HID + 255)/256, 256, 0, stream>>>(gi, gh, sh, y_mask, s_out);
  k_logit<<<(B_*MEMB + 255)/256, 256, 0, stream>>>(ct, s_out, embw, Wct, Wct_b, Wht, Wht_b, logit);
  k_out<<<(B_*128 + 255)/256, 256, 0, stream>>>(logit, W0, W0_b, out);
}
